// PFNNLayer_4930622455811
// MI455X (gfx1250) — compile-verified, matrix operand order checked by reading, not run on hardware
//
#include <hip/hip_runtime.h>
#include <hip/hip_bf16.h>
#include <math.h>

// Problem constants (from reference): B=8192, IN=OUT=1024, 4 knots.
#define B_ROWS   8192
#define IN_DIM   1024
#define OUT_DIM  1024
#define NK       4
#define KP       (NK * IN_DIM)   // 4096 : augmented reduction dim (knot-major)

typedef __bf16 bf16x16 __attribute__((ext_vector_type(16)));
typedef __bf16 bf16x4  __attribute__((ext_vector_type(4)));
typedef float  f32x8   __attribute__((ext_vector_type(8)));
typedef unsigned int u32x4 __attribute__((ext_vector_type(4)));

union Frag { bf16x16 v; u32x4 q[2]; };

// ---------------------------------------------------------------------------
// Prepass 1: A'[b, k*1024+i] = bf16( c[b,k] * feats[b,i] ),  cbuf[b] = c[b,:]
// Cubic (Catmull-Rom) coefficients folded into the activations so the 4 knot
// GEMMs collapse into a single M=8192, N=1024, K=4096 GEMM.
// ---------------------------------------------------------------------------
__global__ void pfnn_prep_act(const float* __restrict__ x,
                              __bf16* __restrict__ abf,
                              float4* __restrict__ cbuf)
{
    const int tid = blockIdx.x * blockDim.x + threadIdx.x;   // b*1024 + i
    const int b = tid >> 10;
    const int i = tid & (IN_DIM - 1);

    const float phase = x[(size_t)b * (IN_DIM + 1) + IN_DIM]; // broadcast (L0 hit)
    const float ps  = (float)NK * phase;
    const float fl  = floorf(ps);
    const float mu  = ps - fl;
    const int   i1  = ((int)fl) & 3;
    const float mu2 = mu * mu, mu3 = mu2 * mu;
    const float w0 = -0.5f*mu3 +      mu2 - 0.5f*mu;
    const float w1 =  1.5f*mu3 - 2.5f*mu2 + 1.0f;
    const float w2 = -1.5f*mu3 + 2.0f*mu2 + 0.5f*mu;
    const float w3 =  0.5f*mu3 - 0.5f*mu2;
    // knot j receives weight w[(j+1-i1) & 3] (idx is a rotation since K=4)
    const float c0 = (i1==0)?w1 : (i1==1)?w0 : (i1==2)?w3 : w2;
    const float c1 = (i1==0)?w2 : (i1==1)?w1 : (i1==2)?w0 : w3;
    const float c2 = (i1==0)?w3 : (i1==1)?w2 : (i1==2)?w1 : w0;
    const float c3 = (i1==0)?w0 : (i1==1)?w3 : (i1==2)?w2 : w1;

    const float f = x[(size_t)b * (IN_DIM + 1) + i];
    __bf16* dst = abf + (size_t)b * KP + i;
    dst[0 * IN_DIM] = (__bf16)(c0 * f);
    dst[1 * IN_DIM] = (__bf16)(c1 * f);
    dst[2 * IN_DIM] = (__bf16)(c2 * f);
    dst[3 * IN_DIM] = (__bf16)(c3 * f);

    if (i == 0) cbuf[b] = make_float4(c0, c1, c2, c3);
}

// ---------------------------------------------------------------------------
// Prepass 2: weight f32 -> bf16 (layout kept (k,o,i); row-major i is exactly
// the per-lane contiguous-K chunks the WMMA B operand wants).
// ---------------------------------------------------------------------------
__global__ void pfnn_prep_w(const float4* __restrict__ w,
                            bf16x4* __restrict__ wbf)
{
    const int t = blockIdx.x * blockDim.x + threadIdx.x;
    const float4 f = w[t];
    bf16x4 o;
    o[0] = (__bf16)f.x; o[1] = (__bf16)f.y;
    o[2] = (__bf16)f.z; o[3] = (__bf16)f.w;
    wbf[t] = o;
}

// ---------------------------------------------------------------------------
// Single fused GEMM + cubic bias blend + ELU.
// WG = 256 threads = 8 waves arranged 2(M) x 4(N); WG tile 128x256.
// Wave tile 64x64 = 4x4 WMMA tiles -> 16 v_wmma per K-step of 32,
// fed by 16 global_load_b128 (1.0 loads/WMMA). All tile strides are folded
// into the 24-bit instruction offsets; only 2 pointer bumps per iteration.
// ---------------------------------------------------------------------------
__global__ __launch_bounds__(256)
void pfnn_wmma_gemm(const __bf16* __restrict__ abf,
                    const __bf16* __restrict__ wbf,
                    const float4* __restrict__ cbuf,
                    const float*  __restrict__ bias,
                    float* __restrict__ out)
{
    const int lane  = threadIdx.x & 31;
    const int wave  = threadIdx.x >> 5;
    const int wm    = wave & 1;               // 0..1  (M)
    const int wn    = wave >> 1;              // 0..3  (N)
    const int mbase = blockIdx.y * 128 + wm * 64;
    const int nbase = blockIdx.x * 256 + wn * 64;
    const int lr    = lane & 15;              // row (A) / col (B) within tile
    const int khalf = (lane >> 4) << 3;       // lanes 0-15: K chunks {0-7,16-23};
                                              // lanes 16-31: {8-15,24-31}
    f32x8 acc[4][4] = {};

    const __bf16* aBase = abf + (size_t)(mbase + lr) * KP + khalf;
    const __bf16* bBase = wbf + (size_t)(nbase + lr) * IN_DIM + khalf;

#pragma unroll 1
    for (int knot = 0; knot < NK; ++knot) {
        const __bf16* aptr = aBase + knot * IN_DIM;               // +2KB/knot
        const __bf16* bptr = bBase + (size_t)knot * OUT_DIM * IN_DIM; // +2MB/knot
#pragma unroll 1
        for (int i0 = 0; i0 < IN_DIM; i0 += 32) {
            Frag a[4], b[4];
#pragma unroll
            for (int t = 0; t < 4; ++t) {
                // A tile stride: 16 rows * KP = 128KB -> immediate offset
                a[t].q[0] = *(const u32x4*)(const void*)(aptr + t * 16 * KP);
                a[t].q[1] = *(const u32x4*)(const void*)(aptr + t * 16 * KP + 16);
            }
#pragma unroll
            for (int s = 0; s < 4; ++s) {
                // B tile stride: 16 rows * IN_DIM = 32KB -> immediate offset
                b[s].q[0] = *(const u32x4*)(const void*)(bptr + s * 16 * IN_DIM);
                b[s].q[1] = *(const u32x4*)(const void*)(bptr + s * 16 * IN_DIM + 16);
            }
#pragma unroll
            for (int s = 0; s < 4; ++s)
#pragma unroll
                for (int t = 0; t < 4; ++t)
                    acc[t][s] = __builtin_amdgcn_wmma_f32_16x16x32_bf16(
                        false, a[t].v, false, b[s].v, (short)0, acc[t][s],
                        false, false);
            aptr += 32;
            bptr += 32;
        }
    }

    // Epilogue: out = ELU( acc + c[b,:] . bias[:,o] )
    float bs[4][4];
#pragma unroll
    for (int s = 0; s < 4; ++s) {
        const int n = nbase + s * 16 + lr;
#pragma unroll
        for (int k = 0; k < 4; ++k)
            bs[s][k] = bias[k * OUT_DIM + n];
    }
    const int mofs = (lane >> 4) << 3;   // C layout: VGPR j holds M=j (+8 for hi lanes)
#pragma unroll
    for (int t = 0; t < 4; ++t) {
#pragma unroll
        for (int j = 0; j < 8; ++j) {
            const int m = mbase + t * 16 + j + mofs;
            const float4 c = cbuf[m];
#pragma unroll
            for (int s = 0; s < 4; ++s) {
                float v = acc[t][s][j]
                        + c.x * bs[s][0] + c.y * bs[s][1]
                        + c.z * bs[s][2] + c.w * bs[s][3];
                v = (v > 0.0f) ? v : (expf(v) - 1.0f);
                out[(size_t)m * OUT_DIM + nbase + s * 16 + lr] = v;
            }
        }
    }
}

// ---------------------------------------------------------------------------
extern "C" void kernel_launch(void* const* d_in, const int* in_sizes, int n_in,
                              void* d_out, int out_size, void* d_ws, size_t ws_size,
                              hipStream_t stream)
{
    (void)in_sizes; (void)n_in; (void)out_size; (void)ws_size;

    const float* x  = (const float*)d_in[0];   // (8192, 1025)
    const float* wk = (const float*)d_in[1];   // (4, 1024, 1024)
    const float* bk = (const float*)d_in[2];   // (4, 1024)
    float* out = (float*)d_out;                // (8192, 1024)

    // workspace: A' bf16 (67.1MB) | W bf16 (8.4MB) | c float4 (128KB)
    const size_t abf_bytes = (size_t)B_ROWS * KP * 2;
    const size_t wbf_bytes = (size_t)NK * OUT_DIM * IN_DIM * 2;
    __bf16* abf  = (__bf16*)d_ws;
    __bf16* wbf  = (__bf16*)((char*)d_ws + abf_bytes);
    float4* cbuf = (float4*)((char*)d_ws + abf_bytes + wbf_bytes);

    pfnn_prep_act<<<dim3((B_ROWS * IN_DIM) / 256), dim3(256), 0, stream>>>(x, abf, cbuf);
    pfnn_prep_w<<<dim3((NK * OUT_DIM * IN_DIM / 4) / 256), dim3(256), 0, stream>>>(
        (const float4*)wk, (bf16x4*)wbf);
    pfnn_wmma_gemm<<<dim3(OUT_DIM / 256, B_ROWS / 128), dim3(256), 0, stream>>>(
        abf, wbf, cbuf, bk, out);
}
